// KascadeReuseAttention_28312424415933
// MI455X (gfx1250) — matrix operand order repaired and checked by reading, not compiled, not run on hardware
//
#include <hip/hip_runtime.h>
#include <hip/hip_bf16.h>
#include <stdint.h>

// ---------------------------------------------------------------------------
// Types for WMMA bf16 path (CDNA5: V_WMMA_F32_16X16X32_BF16, wave32)
// ---------------------------------------------------------------------------
typedef __attribute__((ext_vector_type(16))) __bf16 v16bf;
typedef __attribute__((ext_vector_type(8)))  float  v8f;

union Frag16 { v16bf v; uint32_t u[8]; };
union Acc8   { v8f  v; float   f[8]; };

// CDNA5 async global->LDS path (ASYNCcnt) via inline asm; sync fallback else.
#if defined(__gfx1250__)
#define USE_ASYNC_LDS 1
#else
#define USE_ASYNC_LDS 0
#endif

#if USE_ASYNC_LDS
// GLOBAL_LOAD_ASYNC_TO_LDS_B128: per-lane 16 B global->LDS copy, tracked by ASYNCcnt
__device__ __forceinline__ void async_load_b128(const void* gaddr, void* ldsaddr) {
  const uint32_t l = (uint32_t)(uintptr_t)ldsaddr;   // LDS offset = low 32 bits
  const uint64_t g = (uint64_t)(uintptr_t)gaddr;
  asm volatile("global_load_async_to_lds_b128 %0, %1, off"
               :: "v"(l), "v"(g) : "memory");
}
__device__ __forceinline__ void wait_async_lds() {
  asm volatile("s_wait_asynccnt 0x0" ::: "memory");
}
#else
__device__ __forceinline__ void wait_async_lds() {}
#endif

__device__ __forceinline__ uint16_t f2bf(float x) {
  uint32_t u = __float_as_uint(x);
  uint32_t r = u + 0x7FFFu + ((u >> 16) & 1u);   // round-to-nearest-even
  return (uint16_t)(r >> 16);
}

__device__ __forceinline__ float2 bf2x(uint32_t u) {
  float2 r;
  r.x = __uint_as_float(u << 16);
  r.y = __uint_as_float(u & 0xFFFF0000u);
  return r;
}

// ---------------------------------------------------------------------------
// Elementwise f32 -> bf16
// ---------------------------------------------------------------------------
__global__ void cvt_f32_bf16(const float* __restrict__ in,
                             uint16_t* __restrict__ out, int n) {
  int i = blockIdx.x * blockDim.x + threadIdx.x;
  if (i < n) out[i] = f2bf(in[i]);
}

// Pack Wq|Wk|Wv (each 1024x1024 row-major) into one bf16 B matrix [1024][3072]
__global__ void pack_wqkv(const float* __restrict__ Wq,
                          const float* __restrict__ Wk,
                          const float* __restrict__ Wv,
                          uint16_t* __restrict__ out) {
  int i = blockIdx.x * blockDim.x + threadIdx.x;   // over 1024*3072
  if (i >= 1024 * 3072) return;
  int n = i % 3072, k = i / 3072;
  float v;
  if (n < 1024)       v = Wq[k * 1024 + n];
  else if (n < 2048)  v = Wk[k * 1024 + (n - 1024)];
  else                v = Wv[k * 1024 + (n - 2048)];
  out[i] = f2bf(v);
}

// ---------------------------------------------------------------------------
// bf16 WMMA GEMM: C[M][N] (f32) = A[M][K] (bf16, row-major) @ B[K][N] (bf16)
// Block tile 128x64, 8 waves (4x2), wave tile 32x32 = 2x2 WMMA 16x16x32.
// Double-buffered LDS; A staged via GLOBAL_LOAD_ASYNC_TO_LDS_B128 (ASYNCcnt),
// B staged through registers with an LDS transpose to [n][k].
// ---------------------------------------------------------------------------
#define BM 128
#define BN 64
#define BK 32

__global__ void __launch_bounds__(256)
gemm_bf16_wmma(const uint16_t* __restrict__ A, const uint16_t* __restrict__ B,
               float* __restrict__ C, int M, int N, int K) {
  __shared__ uint16_t As[2][BM][BK];   // [buf][m][k]  16 KB
  __shared__ uint16_t Bs[2][BN][BK];   // [buf][n][k]   8 KB (K-pairs contiguous)

  const int tid  = threadIdx.x;
  const int lane = tid & 31;
  const int wid  = tid >> 5;
  const int wm   = wid & 3;         // 0..3  (M direction)
  const int wn   = wid >> 2;        // 0..1  (N direction)
  const int m0   = blockIdx.x * BM;
  const int n0   = blockIdx.y * BN;
  const int r    = lane & 15;       // row/col within 16
  const int hf   = lane >> 4;       // half-wave select

  // Staging assignments (loop-invariant)
  const int aRow = tid >> 1;            // 0..127
  const int aOff = (tid & 1) * 16;      // 0 / 16
  const int bRow = tid >> 3;            // 0..31
  const int bCol = (tid & 7) * 8;       // 0..56

  Acc8 acc[2][2];
#pragma unroll
  for (int i = 0; i < 2; i++)
#pragma unroll
    for (int j = 0; j < 2; j++)
#pragma unroll
      for (int e = 0; e < 8; e++) acc[i][j].f[e] = 0.f;

  auto issueA = [&](int buf, int k0) {
    const uint16_t* gp = A + (size_t)(m0 + aRow) * K + k0 + aOff;
#if USE_ASYNC_LDS
    async_load_b128(gp,     &As[buf][aRow][aOff]);
    async_load_b128(gp + 8, &As[buf][aRow][aOff + 8]);
#else
    uint4 d0 = *(const uint4*)gp;
    uint4 d1 = *(const uint4*)(gp + 8);
    *(uint4*)&As[buf][aRow][aOff]     = d0;
    *(uint4*)&As[buf][aRow][aOff + 8] = d1;
#endif
  };

  auto loadB = [&](int k0) -> uint4 {
    return *(const uint4*)(B + (size_t)(k0 + bRow) * N + n0 + bCol);
  };
  auto storeB = [&](int buf, uint4 d) {
    const uint16_t* sv = (const uint16_t*)&d;
#pragma unroll
    for (int e = 0; e < 8; e++) Bs[buf][bCol + e][bRow] = sv[e];
  };

  // ---- prologue: stage tile 0 ----
  issueA(0, 0);
  uint4 breg = loadB(0);
  storeB(0, breg);
  wait_async_lds();
  __syncthreads();

  for (int k0 = 0; k0 < K; k0 += BK) {
    const int  cur     = (k0 >> 5) & 1;
    const int  nxt     = cur ^ 1;
    const bool hasNext = (k0 + BK) < K;

    if (hasNext) {               // overlap next-tile movement with WMMAs
      issueA(nxt, k0 + BK);
      breg = loadB(k0 + BK);
    }

    // --- build fragments per CDNA5 16-bit A/B VGPR layouts ---
    Frag16 afrag[2], bfrag[2];
#pragma unroll
    for (int sub = 0; sub < 2; sub++) {
      const int row = wm * 32 + sub * 16 + r;
#pragma unroll
      for (int i = 0; i < 8; i++) {
        // A 16x32: lanes0-15 K=0..7 in v0-3, K=16..23 in v4-7; lanes16-31 +8
        const int kb = (i < 4) ? (2 * i) : (16 + 2 * (i - 4));
        afrag[sub].u[i] = *(const uint32_t*)&As[cur][row][kb + 8 * hf];
      }
    }
#pragma unroll
    for (int sub = 0; sub < 2; sub++) {
      const int col = wn * 32 + sub * 16 + r;
#pragma unroll
      for (int i = 0; i < 8; i++) {
        // B 32x16: lanes0-15 K=0..7 in v0-3, K=16..23 in v4-7; lanes16-31 +8
        bfrag[sub].u[i] = *(const uint32_t*)&Bs[cur][col][((i < 4) ? (2 * i) : (16 + 2 * (i - 4))) + 8 * hf];
      }
    }

#pragma unroll
    for (int i = 0; i < 2; i++)
#pragma unroll
      for (int j = 0; j < 2; j++)
        acc[i][j].v = __builtin_amdgcn_wmma_f32_16x16x32_bf16(
            false, afrag[i].v, false, bfrag[j].v,
            (short)0, acc[i][j].v, false, false);

    if (hasNext) storeB(nxt, breg);   // nxt buffer free since last barrier
    wait_async_lds();                 // A(nxt) resident in LDS
    __syncthreads();
  }

  // --- store: C/D layout: v_e -> M = e + 8*hf, N = lane%16 ---
#pragma unroll
  for (int i = 0; i < 2; i++)
#pragma unroll
    for (int j = 0; j < 2; j++) {
      const int col = n0 + wn * 32 + j * 16 + r;
#pragma unroll
      for (int e = 0; e < 8; e++) {
        const int row = m0 + wm * 32 + i * 16 + hf * 8 + e;
        C[(size_t)row * N + col] = acc[i][j].f[e];
      }
    }
}

// ---------------------------------------------------------------------------
// RoPE (half-split) + layout transform [S][3072] -> packed-bf16 Q/K/V [H][S][64]
// Each thread produces one packed pair (dims 2p, 2p+1).
// ---------------------------------------------------------------------------
__global__ void rope_split_bf16(const float* __restrict__ QKV,
                                const float* __restrict__ cosT,
                                const float* __restrict__ sinT,
                                uint32_t* __restrict__ Qb,
                                uint32_t* __restrict__ Kb,
                                uint32_t* __restrict__ Vb) {
  int idx = blockIdx.x * blockDim.x + threadIdx.x;     // S*H*32 = 1M
  if (idx >= 2048 * 16 * 32) return;
  const int p = idx & 31;          // pair -> dims 2p, 2p+1
  const int h = (idx >> 5) & 15;
  const int s = idx >> 9;

  const float* base = QKV + (size_t)s * 3072 + h * 64;

  float qv[2], kv[2], vv[2];
#pragma unroll
  for (int e = 0; e < 2; e++) {
    const int d   = 2 * p + e;
    const float c  = cosT[s * 32 + (d & 31)];
    const float sn = sinT[s * 32 + (d & 31)];
    const int   dr  = (d < 32) ? d + 32 : d - 32;      // rotate_half partner
    const float sgn = (d < 32) ? -1.f : 1.f;
    qv[e] = base[d] * c        + sgn * base[dr] * sn;
    kv[e] = base[1024 + d] * c + sgn * base[1024 + dr] * sn;
    vv[e] = base[2048 + d];
  }

  const size_t o = ((size_t)h * 2048 + s) * 32 + p;
  Qb[o] = (uint32_t)f2bf(qv[0]) | ((uint32_t)f2bf(qv[1]) << 16);
  Kb[o] = (uint32_t)f2bf(kv[0]) | ((uint32_t)f2bf(kv[1]) << 16);
  Vb[o] = (uint32_t)f2bf(vv[0]) | ((uint32_t)f2bf(vv[1]) << 16);
}

// ---------------------------------------------------------------------------
// Sparse (anchor-tile) attention: one wave32 per (h, query).
// 64 gathered keys = 3 anchor tiles + local tile, 16 tokens each.
// Q/K/V are packed bf16 (halves gather traffic); math in f32.
// Output bf16 [S][H*64] ready as A-matrix for the Wo GEMM.
// ---------------------------------------------------------------------------
__global__ void __launch_bounds__(256)
sparse_attn(const uint32_t* __restrict__ Qb, const uint32_t* __restrict__ Kb,
            const uint32_t* __restrict__ Vb, const int* __restrict__ anchors,
            uint16_t* __restrict__ out) {
  const int lane = threadIdx.x & 31;
  const int wid  = blockIdx.x * (blockDim.x >> 5) + (threadIdx.x >> 5);
  if (wid >= 16 * 2048) return;
  const int h = wid >> 11;
  const int s = wid & 2047;

  int tiles[4];
  {
    const int* ap = anchors + ((size_t)h * 2048 + s) * 3;
    tiles[0] = ap[0]; tiles[1] = ap[1]; tiles[2] = ap[2]; tiles[3] = s >> 4;
  }

  const float2 q = bf2x(Qb[((size_t)h * 2048 + s) * 32 + lane]);

  // ---- logits: lane owns keys 2*lane and 2*lane+1 ----
  float la = 0.f, lb = 0.f;
  for (int j = 0; j < 64; j++) {
    int key = tiles[j >> 4] * 16 + (j & 15);
    key = (key < 0) ? 0 : ((key > 2047) ? 2047 : key);
    const float2 kv = bf2x(Kb[((size_t)h * 2048 + key) * 32 + lane]);
    float p = q.x * kv.x + q.y * kv.y;
#pragma unroll
    for (int o = 16; o > 0; o >>= 1) p += __shfl_xor(p, o, 32);
    float lg = p * 0.125f;                  // 1/sqrt(64)
    if (key > s) lg = -1e10f;               // causal mask on gathered keys
    if ((j >> 1) == lane) { if (j & 1) lb = lg; else la = lg; }
  }

  // ---- softmax over 64 in-register logits ----
  float m = fmaxf(la, lb);
#pragma unroll
  for (int o = 16; o > 0; o >>= 1) m = fmaxf(m, __shfl_xor(m, o, 32));
  float ea = __expf(la - m), eb = __expf(lb - m);
  float sum = ea + eb;
#pragma unroll
  for (int o = 16; o > 0; o >>= 1) sum += __shfl_xor(sum, o, 32);
  const float inv = (sum > 0.f) ? (1.f / sum) : 0.f;
  ea *= inv; eb *= inv;

  // ---- weighted V aggregation: lane owns dims 2*lane, 2*lane+1 ----
  float ox = 0.f, oy = 0.f;
  for (int j = 0; j < 64; j++) {
    int key = tiles[j >> 4] * 16 + (j & 15);
    key = (key < 0) ? 0 : ((key > 2047) ? 2047 : key);
    const float w = __shfl((j & 1) ? eb : ea, j >> 1, 32);
    const float2 vv = bf2x(Vb[((size_t)h * 2048 + key) * 32 + lane]);
    ox += w * vv.x; oy += w * vv.y;
  }

  const uint32_t packed = (uint32_t)f2bf(ox) | ((uint32_t)f2bf(oy) << 16);
  *(uint32_t*)&out[(size_t)s * 1024 + h * 64 + lane * 2] = packed;
}

// ---------------------------------------------------------------------------
// Host-side orchestration
// ---------------------------------------------------------------------------
extern "C" void kernel_launch(void* const* d_in, const int* in_sizes, int n_in,
                              void* d_out, int out_size, void* d_ws, size_t ws_size,
                              hipStream_t stream) {
  (void)in_sizes; (void)n_in; (void)out_size; (void)ws_size;
  const float* x    = (const float*)d_in[0];
  const float* Wq   = (const float*)d_in[1];
  const float* Wk   = (const float*)d_in[2];
  const float* Wv   = (const float*)d_in[3];
  const float* Wo   = (const float*)d_in[4];
  const float* cosT = (const float*)d_in[5];
  const float* sinT = (const float*)d_in[6];
  const int*   anc  = (const int*)d_in[7];
  float* outp = (float*)d_out;

  uint8_t* ws = (uint8_t*)d_ws;
  uint16_t* Xb    = (uint16_t*)(ws);                         //  4 MB  x bf16
  uint16_t* Wqkvb = (uint16_t*)(ws + ((size_t)4  << 20));    //  6 MB  [Wq|Wk|Wv]
  uint16_t* Wob   = (uint16_t*)(ws + ((size_t)10 << 20));    //  2 MB  Wo bf16
  float*    QKV   = (float*)   (ws + ((size_t)12 << 20));    // 24 MB  [S][3072]
  uint32_t* Qb    = (uint32_t*)(ws + ((size_t)36 << 20));    //  4 MB  packed bf16
  uint32_t* Kb    = (uint32_t*)(ws + ((size_t)40 << 20));    //  4 MB
  uint32_t* Vb    = (uint32_t*)(ws + ((size_t)44 << 20));    //  4 MB
  uint16_t* attnb = (uint16_t*)(ws + ((size_t)48 << 20));    //  4 MB  [S][1024]

  // 1) precision conversion / packing
  cvt_f32_bf16<<<(2048 * 1024 + 255) / 256, 256, 0, stream>>>(x, Xb, 2048 * 1024);
  pack_wqkv<<<(1024 * 3072 + 255) / 256, 256, 0, stream>>>(Wq, Wk, Wv, Wqkvb);
  cvt_f32_bf16<<<(1024 * 1024 + 255) / 256, 256, 0, stream>>>(Wo, Wob, 1024 * 1024);

  // 2) fused QKV projection: [2048,1024] @ [1024,3072] via bf16 WMMA
  {
    dim3 grid(2048 / BM, 3072 / BN);
    gemm_bf16_wmma<<<grid, 256, 0, stream>>>(Xb, Wqkvb, QKV, 2048, 3072, 1024);
  }

  // 3) RoPE + head-major packed-bf16 layout
  rope_split_bf16<<<(2048 * 16 * 32) / 256, 256, 0, stream>>>(QKV, cosT, sinT,
                                                              Qb, Kb, Vb);

  // 4) anchor-tile sparse attention (wave32 per query-head)
  sparse_attn<<<(16 * 2048) / 8, 256, 0, stream>>>(Qb, Kb, Vb, anc, attnb);

  // 5) output projection: [2048,1024] @ [1024,1024] via bf16 WMMA -> f32 out
  {
    dim3 grid(2048 / BM, 1024 / BN);
    gemm_bf16_wmma<<<grid, 256, 0, stream>>>(attnb, Wob, outp, 2048, 1024, 1024);
  }
}
